// CausalSelfAttention_73873437491414
// MI455X (gfx1250) — matrix operand order repaired and checked by reading, not compiled, not run on hardware
//
#include <hip/hip_runtime.h>
#include <cmath>
#include <cstdint>

// ---------------- types ----------------
typedef __bf16 bf16_t;
typedef __bf16 bf16x8 __attribute__((ext_vector_type(8)));
typedef __bf16 v16bf  __attribute__((ext_vector_type(16)));
typedef float  v8f    __attribute__((ext_vector_type(8)));

#define B_   2
#define T_   2048
#define C_   1024
#define H_   16
#define DH_  64
#define QKVC (3 * C_)

union FragU { v16bf v; bf16x8 h[2]; };

// A-fragment (16xK=32, 16-bit): lane = M row (0..15), lane>>4 selects K-halves.
// element e -> K = (e>>3)*16 + khalf*8 + (e&7)  => two contiguous 8-elem (16B) chunks.
static __device__ __forceinline__ v16bf load_fragA(const bf16_t* p, int ld, int lane) {
  const int mrow = lane & 15, khalf = lane >> 4;
  const bf16_t* r = p + (size_t)mrow * ld + khalf * 8;
  FragU u;
  u.h[0] = *(const bf16x8*)(r);
  u.h[1] = *(const bf16x8*)(r + 16);
  return u.v;
}

// B-fragment (K=32 x 16, 16-bit) from B^T stored (N,K) row-major:
// lane = N row, element e -> K = (e>>3)*16 + khalf*8 + (e&7)  => two contiguous 8-elem chunks.
static __device__ __forceinline__ v16bf load_fragB(const bf16_t* p, int ld, int lane) {
  const int nrow = lane & 15, khalf = lane >> 4;
  const bf16_t* r = p + (size_t)nrow * ld + khalf * 8;
  FragU u;
  u.h[0] = *(const bf16x8*)(r);
  u.h[1] = *(const bf16x8*)(r + 16);
  return u.v;
}

static __device__ __forceinline__ v8f wmma_bf16(v16bf a, v16bf b, v8f c) {
  return __builtin_amdgcn_wmma_f32_16x16x32_bf16(false, a, false, b, (short)0, c,
                                                 false, false);
}

// async DMA of one 16B chunk: global -> LDS (ASYNCcnt-tracked, gfx1250)
static __device__ __forceinline__ void async_copy_b128(const bf16_t* gsrc, bf16_t* ldst) {
  const unsigned loff = (unsigned)(uintptr_t)ldst;   // LDS byte offset (addr[31:0])
  asm volatile("global_load_async_to_lds_b128 %0, %1, off"
               :: "v"(loff), "v"(gsrc) : "memory");
}

// ---------------- conversion kernels ----------------
__global__ void k_f32_to_bf16(const float* __restrict__ src, bf16_t* __restrict__ dst, int n) {
  int i = blockIdx.x * blockDim.x + threadIdx.x;
  if (i < n) dst[i] = (bf16_t)src[i];
}

// src (K,N) fp32 row-major -> dst (N,K) bf16 row-major
__global__ void k_transpose_to_bf16(const float* __restrict__ src, bf16_t* __restrict__ dst,
                                    int K, int N) {
  int i = blockIdx.x * blockDim.x + threadIdx.x;
  if (i >= K * N) return;
  int n = i % N, k = i / N;
  dst[(size_t)n * K + k] = (bf16_t)src[i];
}

// ---------------- tiled bf16 WMMA GEMM with async-LDS double buffering ----------------
//  C(M,N) = A(M,K) * B(K,N), B given as B^T (N,K).  Block tile 128x64, 8 waves (4Mx2N),
//  K-step 32.  A/B K-slices staged in LDS via global_load_async_to_lds_b128.
template <bool STORE_F32>
__global__ __launch_bounds__(256)
void k_gemm_bf16(const bf16_t* __restrict__ A, const bf16_t* __restrict__ BT,
                 void* __restrict__ Cout, int M, int N, int K) {
  __shared__ __align__(16) bf16_t tA[2][128 * 32];   // 2 x 8KB
  __shared__ __align__(16) bf16_t tB[2][64 * 32];    // 2 x 4KB
  const int tid  = threadIdx.x;
  const int lane = tid & 31;
  const int wave = tid >> 5;
  const int bm = blockIdx.y * 128, bn = blockIdx.x * 64;
  const int wm0 = (wave >> 1) * 32, wn0 = (wave & 1) * 32;

  // Issue the async batch for K-slice k0 into buffer `buf`:
  //   A tile: 128 rows x 32 cols = 512 x 16B chunks -> 2 per thread
  //   B tile:  64 rows x 32 cols = 256 x 16B chunks -> 1 per thread
  auto issue = [&](int buf, int k0) {
    #pragma unroll
    for (int i = 0; i < 2; ++i) {
      const int c = tid + 256 * i;
      const int row = c >> 2, off = (c & 3) * 8;
      async_copy_b128(A + (size_t)(bm + row) * K + k0 + off, &tA[buf][row * 32 + off]);
    }
    {
      const int row = tid >> 2, off = (tid & 3) * 8;
      async_copy_b128(BT + (size_t)(bn + row) * K + k0 + off, &tB[buf][row * 32 + off]);
    }
  };

  v8f c[2][2] = {};
  issue(0, 0);
  for (int k0 = 0; k0 < K; k0 += 32) {
    const int buf = (k0 >> 5) & 1;
    if (k0 + 32 < K) {
      issue(buf ^ 1, k0 + 32);                       // overlap DMA with this step's math
      asm volatile("s_wait_asynccnt 3" ::: "memory");// in-order: current batch resident
    } else {
      asm volatile("s_wait_asynccnt 0" ::: "memory");
    }
    __syncthreads();                                 // LDS visible to all waves
    v16bf a0 = load_fragA(&tA[buf][wm0 * 32], 32, lane);
    v16bf a1 = load_fragA(&tA[buf][(wm0 + 16) * 32], 32, lane);
    v16bf b0 = load_fragB(&tB[buf][wn0 * 32], 32, lane);
    v16bf b1 = load_fragB(&tB[buf][(wn0 + 16) * 32], 32, lane);
    c[0][0] = wmma_bf16(a0, b0, c[0][0]);
    c[0][1] = wmma_bf16(a0, b1, c[0][1]);
    c[1][0] = wmma_bf16(a1, b0, c[1][0]);
    c[1][1] = wmma_bf16(a1, b1, c[1][1]);
    __syncthreads();                                 // done reading before buffer reuse
  }
  const int hi = lane >> 4, col = lane & 15;         // C layout: row = r + 8*hi, col = lane&15
  #pragma unroll
  for (int i = 0; i < 2; ++i)
    #pragma unroll
    for (int j = 0; j < 2; ++j)
      #pragma unroll
      for (int r = 0; r < 8; ++r) {
        const int row = bm + wm0 + 16 * i + r + 8 * hi;
        const int cc  = bn + wn0 + 16 * j + col;
        if (STORE_F32) ((float*)Cout)[(size_t)row * N + cc] = c[i][j][r];
        else           ((bf16_t*)Cout)[(size_t)row * N + cc] = (bf16_t)c[i][j][r];
      }
}

// ---------------- RoPE + head-major reorder (V stored transposed per head) ----------------
__global__ void k_rope_reorder(const bf16_t* __restrict__ qkv,
                               bf16_t* __restrict__ qh, bf16_t* __restrict__ kh,
                               bf16_t* __restrict__ vT) {
  int gid = blockIdx.x * blockDim.x + threadIdx.x;
  if (gid >= B_ * T_ * H_ * 32) return;
  const int d = gid & 31;
  const int h = (gid >> 5) & 15;
  const int t = (gid >> 9) & (T_ - 1);
  const int b = gid >> 20;
  const size_t rb = (size_t)(b * T_ + t) * QKVC;
  const int co = h * DH_ + d;
  const float q1 = (float)qkv[rb + co],            q2 = (float)qkv[rb + co + 32];
  const float k1 = (float)qkv[rb + C_ + co],       k2 = (float)qkv[rb + C_ + co + 32];
  const float v1 = (float)qkv[rb + 2 * C_ + co],   v2 = (float)qkv[rb + 2 * C_ + co + 32];
  // inv_freq = 10000^(-d/32); angle = t * inv_freq
  const float freq = __expf(-(float)d * (9.210340371976184f / 32.0f));
  const float ang  = (float)t * freq;
  const float cs = __cosf(ang), sn = __sinf(ang);
  const float q1r = q1 * cs - q2 * sn, q2r = q2 * cs + q1 * sn;
  const float k1r = k1 * cs - k2 * sn, k2r = k2 * cs + k1 * sn;
  const int bh = b * H_ + h;
  const size_t ob = ((size_t)bh * T_ + t) * DH_ + d;   // (B,H,T,Dh)
  qh[ob] = (bf16_t)q1r; qh[ob + 32] = (bf16_t)q2r;
  kh[ob] = (bf16_t)k1r; kh[ob + 32] = (bf16_t)k2r;
  const size_t vb = ((size_t)bh * DH_ + d) * T_ + t;   // (B,H,Dh,T)
  vT[vb] = (bf16_t)v1; vT[vb + (size_t)32 * T_] = (bf16_t)v2;
}

// ---------------- flash attention: 1 wave = 16 query rows, 32 keys/step ----------------
__global__ __launch_bounds__(128)
void k_flash_attn(const bf16_t* __restrict__ qh, const bf16_t* __restrict__ kh,
                  const bf16_t* __restrict__ vT, bf16_t* __restrict__ y) {
  __shared__ __align__(16) bf16_t ldsP[4][16 * 32];
  const int lane = threadIdx.x & 31;
  const int wave = threadIdx.x >> 5;
  const int qblk = blockIdx.x & 31;
  const int bh   = blockIdx.x >> 5;
  const int b = bh >> 4, h = bh & 15;
  const int q0 = qblk * 64 + wave * 16;
  const bf16_t* Q  = qh + (size_t)bh * T_ * DH_;
  const bf16_t* Kb = kh + (size_t)bh * T_ * DH_;
  const bf16_t* V  = vT + (size_t)bh * DH_ * T_;

  const v16bf qa0 = load_fragA(Q + (size_t)q0 * DH_, DH_, lane);        // dh 0..31
  const v16bf qa1 = load_fragA(Q + (size_t)q0 * DH_ + 32, DH_, lane);   // dh 32..63

  const int hi = lane >> 4, col = lane & 15;
  float m[8], l[8];
  v8f o[4] = {};
  #pragma unroll
  for (int r = 0; r < 8; ++r) { m[r] = -INFINITY; l[r] = 0.0f; }

  const int nsteps = (q0 + 16 + 31) >> 5;
  for (int s = 0; s < nsteps; ++s) {
    const int k0 = s * 32;
    if (s + 1 < nsteps) {                            // gfx1250 global_prefetch_b8
      __builtin_prefetch(Kb + (size_t)(k0 + 32 + (lane & 15)) * DH_, 0, 0);
      __builtin_prefetch(V + (size_t)(lane & 15) * T_ + k0 + 32, 0, 0);
    }
    // S = Q * K^T  (two 16x16 tiles, K-dim = Dh = 64 chained as 2x32)
    v8f s0 = {}, s1 = {};
    {
      v16bf kb0  = load_fragB(Kb + (size_t)k0 * DH_, DH_, lane);
      v16bf kb0b = load_fragB(Kb + (size_t)k0 * DH_ + 32, DH_, lane);
      s0 = wmma_bf16(qa0, kb0, s0);
      s0 = wmma_bf16(qa1, kb0b, s0);
      v16bf kb1  = load_fragB(Kb + (size_t)(k0 + 16) * DH_, DH_, lane);
      v16bf kb1b = load_fragB(Kb + (size_t)(k0 + 16) * DH_ + 32, DH_, lane);
      s1 = wmma_bf16(qa0, kb1, s1);
      s1 = wmma_bf16(qa1, kb1b, s1);
    }
    // online softmax (row stats live per-lane for rows r + 8*hi, replicated across 16 lanes)
    float alpha[8], p0[8], p1[8];
    #pragma unroll
    for (int r = 0; r < 8; ++r) {
      const int qg = q0 + r + 8 * hi;
      float v0 = s0[r] * 0.125f;            // 1/sqrt(64)
      float v1 = s1[r] * 0.125f;
      if (k0 + col > qg)      v0 = -INFINITY;  // causal mask
      if (k0 + 16 + col > qg) v1 = -INFINITY;
      float mx = fmaxf(v0, v1);
      #pragma unroll
      for (int off = 1; off < 16; off <<= 1)
        mx = fmaxf(mx, __shfl_xor(mx, off, 32));
      const float mnew = fmaxf(m[r], mx);
      alpha[r] = __expf(m[r] - mnew);
      const float e0 = __expf(v0 - mnew);
      const float e1 = __expf(v1 - mnew);
      float rs = e0 + e1;
      #pragma unroll
      for (int off = 1; off < 16; off <<= 1)
        rs += __shfl_xor(rs, off, 32);
      l[r] = l[r] * alpha[r] + rs;
      m[r] = mnew;
      p0[r] = e0; p1[r] = e1;
    }
    #pragma unroll
    for (int cN = 0; cN < 4; ++cN)
      #pragma unroll
      for (int r = 0; r < 8; ++r)
        o[cN][r] *= alpha[r];
    // C-layout P -> A-fragment layout via LDS round trip (per-wave 1KB tile)
    #pragma unroll
    for (int r = 0; r < 8; ++r) {
      ldsP[wave][(r + 8 * hi) * 32 + col]      = (bf16_t)p0[r];
      ldsP[wave][(r + 8 * hi) * 32 + 16 + col] = (bf16_t)p1[r];
    }
    __builtin_amdgcn_wave_barrier();
    const v16bf pa = load_fragA(&ldsP[wave][0], 32, lane);
    // O += P * V  (B from V^T: N = dh, K = key, contiguous rows)
    #pragma unroll
    for (int cN = 0; cN < 4; ++cN) {
      v16bf vb = load_fragB(V + (size_t)(cN * 16) * T_ + k0, T_, lane);
      o[cN] = wmma_bf16(pa, vb, o[cN]);
    }
    __builtin_amdgcn_wave_barrier();
  }
  float linv[8];
  #pragma unroll
  for (int r = 0; r < 8; ++r) linv[r] = 1.0f / l[r];
  #pragma unroll
  for (int cN = 0; cN < 4; ++cN)
    #pragma unroll
    for (int r = 0; r < 8; ++r) {
      const int t = q0 + r + 8 * hi;
      y[(size_t)(b * T_ + t) * C_ + h * DH_ + cN * 16 + col] = (bf16_t)(o[cN][r] * linv[r]);
    }
}

// ---------------- host launcher ----------------
extern "C" void kernel_launch(void* const* d_in, const int* in_sizes, int n_in,
                              void* d_out, int out_size, void* d_ws, size_t ws_size,
                              hipStream_t stream) {
  (void)in_sizes; (void)n_in; (void)out_size; (void)ws_size;
  const float* x     = (const float*)d_in[0];
  const float* Wqkv  = (const float*)d_in[1];
  const float* Wproj = (const float*)d_in[2];
  char* ws = (char*)d_ws;
  // workspace layout (72 MB total)
  bf16_t* xb     = (bf16_t*)(ws + (size_t)0);            //  8 MB  x bf16 (B*T, C)
  bf16_t* wqkvT  = (bf16_t*)(ws + ((size_t)8  << 20));   //  6 MB  W_qkv^T (3C, C)
  bf16_t* wprojT = (bf16_t*)(ws + ((size_t)14 << 20));   //  2 MB  W_proj^T (C, C)
  bf16_t* qkv    = (bf16_t*)(ws + ((size_t)16 << 20));   // 24 MB  qkv bf16 (B*T, 3C)
  bf16_t* qh     = (bf16_t*)(ws + ((size_t)40 << 20));   //  8 MB  (B,H,T,Dh)
  bf16_t* kh     = (bf16_t*)(ws + ((size_t)48 << 20));   //  8 MB  (B,H,T,Dh)
  bf16_t* vT     = (bf16_t*)(ws + ((size_t)56 << 20));   //  8 MB  (B,H,Dh,T)
  bf16_t* yb     = (bf16_t*)(ws + ((size_t)64 << 20));   //  8 MB  y bf16 (B*T, C)

  const int nx = B_ * T_ * C_;
  k_f32_to_bf16<<<(nx + 255) / 256, 256, 0, stream>>>(x, xb, nx);
  k_transpose_to_bf16<<<(C_ * QKVC + 255) / 256, 256, 0, stream>>>(Wqkv, wqkvT, C_, QKVC);
  k_transpose_to_bf16<<<(C_ * C_ + 255) / 256, 256, 0, stream>>>(Wproj, wprojT, C_, C_);
  // QKV projection: (4096 x 3072 x 1024)
  k_gemm_bf16<false><<<dim3(QKVC / 64, (B_ * T_) / 128), 256, 0, stream>>>(
      xb, wqkvT, qkv, B_ * T_, QKVC, C_);
  k_rope_reorder<<<(B_ * T_ * H_ * 32 + 255) / 256, 256, 0, stream>>>(qkv, qh, kh, vT);
  // flash attention: 1024 blocks x 4 waves
  k_flash_attn<<<B_ * H_ * (T_ / 64), 128, 0, stream>>>(qh, kh, vT, yb);
  // output projection: (4096 x 1024 x 1024) -> fp32 d_out
  k_gemm_bf16<true><<<dim3(C_ / 64, (B_ * T_) / 128), 256, 0, stream>>>(
      yb, wprojT, d_out, B_ * T_, C_, C_);
}
